// WaveletTransformLayer_33672543600999
// MI455X (gfx1250) — matrix operand order repaired and checked by reading, not compiled, not run on hardware
//
#include <hip/hip_runtime.h>
#include <hip/hip_bf16.h>
#include <math.h>

typedef _Float16 v8h  __attribute__((ext_vector_type(8)));
typedef _Float16 v16h __attribute__((ext_vector_type(16)));
typedef float    v8f  __attribute__((ext_vector_type(8)));

#define WIN   256
#define L_LEN 16384
#define NCH   8
#define NB    16
#define RDIM  32
#define PI_F  3.14159265358979323846f

#define MWG   512                 // rows per block (8 waves x 64 rows)
#define NRB   4                   // row-blocks (16 rows) per wave

// Shifted-copy stride (halfs): live 767 + pad -> 776 halfs = 1552B = 97*16B
// (16B-aligned); 388 dwords == 4 mod 64 banks -> copies spread all 64 banks.
#define CP_STRIDE 776
#define SIG_SPAN  784             // samples staged: j in [0, 782) rounded up

// ---------------------------------------------------------------------------
// Kernel 1: build the Gabor bank directly in WMMA-B swizzled f16 layout.
//   element[((nt*8 + kc)*32 + lane)*16 + h] = Filt[tau][n]
//   n   = nt*16 + lane%16        (n<32: real, r=n ; n>=32: imag, r=n-32)
//   tau = kc*32 + (lane/16)*16 + h   (B-matrix: lane halfs are 16 contiguous K)
//   filter index f = c*32 + (31 - r)   (the reference's [::-1] flip)
// ---------------------------------------------------------------------------
__global__ __launch_bounds__(256) void gabor_bank_gen(
    const float* __restrict__ kV, const float* __restrict__ sV,
    _Float16* __restrict__ bank) {
  int t = blockIdx.x * 256 + threadIdx.x;      // 131072 total
  int h    =  t        & 15;
  int lane = (t >>  4) & 31;
  int kc   = (t >>  9) & 7;
  int nt   = (t >> 12) & 3;
  int c    =  t >> 14;
  int n    = nt * 16 + (lane & 15);
  int tau  = kc * 32 + (lane >> 4) * 16 + h;
  bool re  = (n < RDIM);
  int  r   = re ? n : n - RDIM;
  int  f   = c * RDIM + (RDIM - 1 - r);
  float kv = kV[f], sv = sV[f];
  float phase = (-2.0f * PI_F / (float)WIN) * kv * (float)tau;
  float d  = (float)tau - 128.0f;                  // mu = W/2
  float g  = rsqrtf(2.0f * PI_F * sv * sv) * expf(-d * d / (2.0f * sv * sv));
  float v  = (re ? cosf(phase) : sinf(phase)) * g;
  bank[t]  = (_Float16)v;
}

// ---------------------------------------------------------------------------
// Kernel 2: main WMMA GEMM. One block = (b, c, 512 rows). 8 waves, wave w
// owns rows [w*64, w*64+64): 4 row-blocks x 4 N-tiles, K=256 in 8 chunks.
// Per K-chunk: 8 A loads + 8 B loads feed 16 WMMAs.
// ---------------------------------------------------------------------------
__global__ __launch_bounds__(256, 4) void wavelet_wmma(
    const float* __restrict__ x, const _Float16* __restrict__ bank,
    float* __restrict__ out) {
  __shared__ _Float16 s_bank[16384];           // 32 KB swizzled B
  __shared__ _Float16 s_cp[16 * CP_STRIDE];    // 16 shifted signal copies

  const int mb  = blockIdx.x;                  // 32 m-blocks
  const int c   = blockIdx.y;                  // 8 channels
  const int b   = blockIdx.z;                  // 16 batches
  const int tid = threadIdx.x;
  const int i0  = mb * MWG;

  // --- stage filter bank (aligned 16B copies; L2-resident after warmup) ---
  {
    const uint4* src = (const uint4*)(bank + (size_t)c * 16384);
    uint4*       dst = (uint4*)s_bank;
    #pragma unroll
    for (int i = 0; i < 8; ++i) dst[tid + 256 * i] = src[tid + 256 * i];
  }
  // --- stage signal straight into the 16 shifted copies -------------------
  // copy m, slot jj holds s[jj+m]; sample j lands at halfword j + m*775.
  // Common case: 16 ds_store_b16 with immediate offsets (m*1550 bytes).
  #pragma unroll
  for (int q = 0; q < 4; ++q) {
    int j = tid + 256 * q;                     // j in [0,1024), used < 784
    if (j >= SIG_SPAN) break;
    int g = i0 - 128 + j;
    float v = (g >= 0 && g < L_LEN) ? x[((size_t)b * L_LEN + g) * NCH + c] : 0.0f;
    _Float16 hv = (_Float16)v;
    if (j >= 15 && j < CP_STRIDE) {
      #pragma unroll
      for (int m = 0; m < 16; ++m) s_cp[j + m * (CP_STRIDE - 1)] = hv;
    } else {
      #pragma unroll
      for (int m = 0; m < 16; ++m) {
        int jj = j - m;
        if (jj >= 0 && jj < CP_STRIDE) s_cp[m * CP_STRIDE + jj] = hv;
      }
    }
  }
  __syncthreads();

  const int lane  = tid & 31;
  const int w     = tid >> 5;
  const int m0    = w * 64;
  const int mlane = lane & 15;    // A row within 16-block / C column
  const int kh    = lane >> 4;    // A half-K selector / C row-group

  v8f acc[NRB][4] = {};           // [row-block][n-tile], 128 VGPRs f32

  #pragma unroll 1
  for (int kc = 0; kc < 8; ++kc) {
    // Fence the scheduler per chunk: <=16 ds loads in flight keeps pressure
    // ~190 VGPRs (no spills, no vgpr-msb).
    __builtin_amdgcn_sched_barrier(0);
    // A tiles: lane = row mlane; halfs {kh*16+0..15} (contiguous K);
    // rb tiles are disjoint 16-half runs -> 8 b128 loads.
    v16h a[NRB];
    #pragma unroll
    for (int rb = 0; rb < NRB; ++rb) {
      const v8h* p = (const v8h*)(s_cp + mlane * CP_STRIDE +
                                  m0 + rb * 16 + kc * 32 + kh * 16);
      v8h lo = p[0];        // K = kc*32 + kh*16 + 0..7
      v8h hi = p[1];        // K = kc*32 + kh*16 + 8..15  (+8 halfs)
      a[rb] = __builtin_shufflevector(lo, hi, 0, 1, 2, 3, 4, 5, 6, 7,
                                              8, 9, 10, 11, 12, 13, 14, 15);
    }
    #pragma unroll
    for (int nt = 0; nt < 4; ++nt) {
      v16h bv = *(const v16h*)(s_bank + (((nt * 8 + kc) * 32 + lane) << 4));
      #pragma unroll
      for (int rb = 0; rb < NRB; ++rb)
        acc[rb][nt] = __builtin_amdgcn_wmma_f32_16x16x32_f16(
            false, a[rb], false, bv, (short)0, acc[rb][nt], false, false);
    }
  }

  // --- power = re^2 + im^2 (per-lane: tiles nt and nt+2 share layout) -----
  // Non-temporal stores: 256MB streamed output must not thrash the 192MB L2
  // that is caching the filter bank and x.
  #pragma unroll
  for (int rb = 0; rb < NRB; ++rb) {
    #pragma unroll
    for (int pt = 0; pt < 2; ++pt) {
      v8f pw = acc[rb][pt] * acc[rb][pt] + acc[rb][pt + 2] * acc[rb][pt + 2];
      int col   = c * RDIM + pt * 16 + mlane;
      int rbase = i0 + m0 + rb * 16 + kh * 8;       // M = vgpr + 8*(lane/16)
      float* op = out + ((size_t)b * L_LEN + rbase) * 256 + col;
      #pragma unroll
      for (int v = 0; v < 8; ++v)
        __builtin_nontemporal_store(pw[v], op + (size_t)v * 256);
    }
  }
}

extern "C" void kernel_launch(void* const* d_in, const int* in_sizes, int n_in,
                              void* d_out, int out_size, void* d_ws, size_t ws_size,
                              hipStream_t stream) {
  const float* x  = (const float*)d_in[0];
  const float* kV = (const float*)d_in[1];
  const float* sV = (const float*)d_in[2];
  float* out      = (float*)d_out;
  _Float16* bank  = (_Float16*)d_ws;     // 131072 halfs = 256 KB scratch

  // 1) build swizzled Gabor bank (8*4*8*32*16 = 131072 elements)
  gabor_bank_gen<<<512, 256, 0, stream>>>(kV, sV, bank);

  // 2) main WMMA transform: grid (m-blocks, channels, batch)
  dim3 grid(L_LEN / MWG, NCH, NB);
  wavelet_wmma<<<grid, 256, 0, stream>>>(x, bank, out);
}
